// ActionThenNodePolicy_63599875719711
// MI455X (gfx1250) — matrix operand order repaired and checked by reading, not compiled, not run on hardware
//
#include <hip/hip_runtime.h>
#include <hip/hip_bf16.h>
#include <math.h>

#define N_NODES 131072
#define N_G     256
#define N_D     256
#define N_A     64
#define NEGV    (-1e9f)
#define EPSV    (1e-10f)

typedef __attribute__((ext_vector_type(16))) __bf16 v16bf;
typedef __attribute__((ext_vector_type(8)))  float  v8f;

union V16U { uint4 q[2]; v16bf v; };

__device__ __forceinline__ unsigned short f2bf(float x) {
  union { float f; unsigned int u; } c; c.f = x;
  unsigned int r = c.u + 0x7fffu + ((c.u >> 16) & 1u);   // RNE
  return (unsigned short)(r >> 16);
}
__device__ __forceinline__ float bf2f(unsigned short h) {
  union { float f; unsigned int u; } c; c.u = ((unsigned int)h) << 16;
  return c.f;
}

// 16x16 output tile, K=256, A from LDS bf16 hi/lo planes [32][256],
// B from global bf16 hi/lo planes [64][256] (row-major [a][d]).
// 3-term bf16 split: Ahi*Bhi + Alo*Bhi + Ahi*Blo  (fp32-class accuracy).
__device__ __forceinline__ v8f wmma_tile_k256(
    const unsigned short* __restrict__ vhi,
    const unsigned short* __restrict__ vlo,
    const unsigned short* __restrict__ whi,
    const unsigned short* __restrict__ wlo,
    int m0, int a0, int lane)
{
  v8f c = {0.f, 0.f, 0.f, 0.f, 0.f, 0.f, 0.f, 0.f};
  const int row  = m0 + (lane & 15);
  const int acol = a0 + (lane & 15);
  const int aoff = (lane & 16) ? 8  : 0;   // A: lanes 16-31 hold K+8 / K+24
  const int boff = (lane & 16) ? 8  : 0;   // B: lanes 16-31 hold K+8 / K+24
  const unsigned short* va = vhi + row  * N_D + aoff;
  const unsigned short* vb = vlo + row  * N_D + aoff;
  const unsigned short* wa = whi + acol * N_D + boff;
  const unsigned short* wb = wlo + acol * N_D + boff;
#pragma unroll
  for (int k0 = 0; k0 < N_D; k0 += 32) {
    V16U Ah, Al, Bh, Bl;
    Ah.q[0] = *(const uint4*)(va + k0);      Ah.q[1] = *(const uint4*)(va + k0 + 16);
    Al.q[0] = *(const uint4*)(vb + k0);      Al.q[1] = *(const uint4*)(vb + k0 + 16);
    Bh.q[0] = *(const uint4*)(wa + k0);      Bh.q[1] = *(const uint4*)(wa + k0 + 16);
    Bl.q[0] = *(const uint4*)(wb + k0);      Bl.q[1] = *(const uint4*)(wb + k0 + 16);
    c = __builtin_amdgcn_wmma_f32_16x16x32_bf16(false, Ah.v, false, Bh.v, (short)0, c, false, false);
    c = __builtin_amdgcn_wmma_f32_16x16x32_bf16(false, Al.v, false, Bh.v, (short)0, c, false, false);
    c = __builtin_amdgcn_wmma_f32_16x16x32_bf16(false, Ah.v, false, Bl.v, (short)0, c, false, false);
  }
  return c;
}

// ---- online softmax helpers (deterministic fixed-order combines) ----
__device__ __forceinline__ void osupd(float& m, float& s, float x) {
  float mn = fmaxf(m, x);
  s = s * expf(m - mn) + expf(x - mn);
  m = mn;
}
__device__ __forceinline__ void oscomb(float& m, float& s, float m2, float s2) {
  if (s2 == 0.f) return;
  if (s == 0.f) { m = m2; s = s2; return; }
  float mn = fmaxf(m, m2);
  s = s * expf(m - mn) + s2 * expf(m2 - mn);
  m = mn;
}

// K0: exclusive prefix scan of n_nodes -> seg_start[G+1]
__global__ void k0_scan(const int* __restrict__ n_nodes, int* __restrict__ seg) {
  if (threadIdx.x == 0 && blockIdx.x == 0) {
    int acc = 0;
    for (int g = 0; g < N_G; ++g) { seg[g] = acc; acc += n_nodes[g]; }
    seg[N_G] = acc;
  }
}

// K0b: split one fp32 weight matrix [64][256] into bf16 hi/lo planes
__global__ __launch_bounds__(256) void k0b_conv(const float* __restrict__ W,
                                                unsigned short* __restrict__ hi,
                                                unsigned short* __restrict__ lo) {
  int i = blockIdx.x * 256 + threadIdx.x;
  if (i < N_A * N_D) {
    float x = W[i];
    unsigned short h = f2bf(x);
    hi[i] = h;
    lo[i] = f2bf(x - bf2f(h));
  }
}

// K1: NGA logits -> out p_n__a region (as scratch) ; node_logits -> ws
__global__ __launch_bounds__(256) void k1_nga_nl(
    const float* __restrict__ values,
    const unsigned short* __restrict__ wnga_hi, const unsigned short* __restrict__ wnga_lo,
    const float* __restrict__ b_nga, const float* __restrict__ w_node,
    float* __restrict__ out_pnna, float* __restrict__ nl_out)
{
  __shared__ __align__(16) unsigned short vhi[32 * N_D];
  __shared__ __align__(16) unsigned short vlo[32 * N_D];
  __shared__ float obuf[32 * N_A];
  const int tid = threadIdx.x;
  const int node0 = blockIdx.x * 32;
#pragma unroll 4
  for (int i = 0; i < 32; ++i) {           // coalesced stage + bf16 split
    float x = values[(size_t)(node0 + i) * N_D + tid];
    unsigned short h = f2bf(x);
    vhi[i * N_D + tid] = h;
    vlo[i * N_D + tid] = f2bf(x - bf2f(h));
  }
  __syncthreads();
  const int wave = tid >> 5, lane = tid & 31;
  const int m0 = (wave >> 2) * 16;         // 8 waves = 2 row-subtiles x 4 col-tiles
  const int a0 = (wave & 3) * 16;
  v8f c = wmma_tile_k256(vhi, vlo, wnga_hi, wnga_lo, m0, a0, lane);
  {
    const int col = a0 + (lane & 15);
    const float bias = b_nga[col];
    const int rbase = m0 + ((lane & 16) ? 8 : 0);
#pragma unroll
    for (int v = 0; v < 8; ++v) obuf[(rbase + v) * N_A + col] = c[v] + bias;
  }
  __syncthreads();
#pragma unroll
  for (int i = 0; i < 8; ++i) {            // contiguous 8KB store
    int e = tid + 256 * i;
    out_pnna[(size_t)node0 * N_A + e] = obuf[e];
  }
  if (tid < 32) {                          // node GEMV (tiny)
    float s = 0.f;
    for (int d = 0; d < N_D; ++d)
      s += (bf2f(vhi[tid * N_D + d]) + bf2f(vlo[tid * N_D + d])) * w_node[d];
    nl_out[node0 + tid] = s;
  }
}

// K2: per-graph online segment max/sum: unmasked & masked NGA per action, node logits
__global__ __launch_bounds__(256) void k2_seg(
    const float* __restrict__ nga, const unsigned char* __restrict__ mask,
    const int* __restrict__ seg_start, const float* __restrict__ nl,
    float* __restrict__ Mm, float* __restrict__ Sm,
    float* __restrict__ Mu, float* __restrict__ Su,
    float* __restrict__ Mn, float* __restrict__ Sn)
{
  const int g = blockIdx.x;
  const int s = seg_start[g], e = seg_start[g + 1];
  const int tid = threadIdx.x;
  const int a = tid & 63, r = tid >> 6;
  const bool mk = mask[g * N_A + a] != 0;
  float mu = -INFINITY, su = 0.f, mm = -INFINITY, sm = 0.f;
  for (int n = s + r; n < e; n += 4) {
    float x = nga[(size_t)n * N_A + a];
    osupd(mu, su, x);
    osupd(mm, sm, mk ? x : NEGV);
  }
  __shared__ float pm[4][64], ps[4][64], qm[4][64], qs[4][64];
  pm[r][a] = mu; ps[r][a] = su; qm[r][a] = mm; qs[r][a] = sm;
  __syncthreads();
  if (tid < 64) {
    float M = -INFINITY, S = 0.f;
    for (int rr = 0; rr < 4; ++rr) oscomb(M, S, pm[rr][tid], ps[rr][tid]);
    Mu[g * N_A + tid] = M; Su[g * N_A + tid] = S;
    M = -INFINITY; S = 0.f;
    for (int rr = 0; rr < 4; ++rr) oscomb(M, S, qm[rr][tid], qs[rr][tid]);
    Mm[g * N_A + tid] = M; Sm[g * N_A + tid] = S;
  }
  float m = -INFINITY, ss = 0.f;
  for (int n = s + tid; n < e; n += 256) osupd(m, ss, nl[n]);
  __shared__ float rm[256], rs[256];
  __syncthreads();
  rm[tid] = m; rs[tid] = ss;
  __syncthreads();
  for (int off = 128; off > 0; off >>= 1) {
    if (tid < off) oscomb(rm[tid], rs[tid], rm[tid + off], rs[tid + off]);
    __syncthreads();
  }
  if (tid == 0) { Mn[g] = rm[0]; Sn[g] = rs[0]; }
}

// K3a: per-graph: recompute AGN via WMMA, p_a = segsum(p_n * softmax(masked AGN))
__global__ __launch_bounds__(256) void k3_agn(
    const float* __restrict__ values,
    const unsigned short* __restrict__ whi, const unsigned short* __restrict__ wlo,
    const float* __restrict__ b_agn, const unsigned char* __restrict__ mask,
    const int* __restrict__ seg_start, const float* __restrict__ nl,
    const float* __restrict__ Mn, const float* __restrict__ Sn,
    float* __restrict__ out_pa)
{
  __shared__ __align__(16) unsigned short vhi[32 * N_D];
  __shared__ __align__(16) unsigned short vlo[32 * N_D];
  __shared__ float abuf[32 * N_A];
  __shared__ float pacc[N_A];
  __shared__ unsigned char smask[N_A];
  const int g = blockIdx.x;
  const int s = seg_start[g], e = seg_start[g + 1];
  const int tid = threadIdx.x;
  if (tid < N_A) { pacc[tid] = 0.f; smask[tid] = mask[g * N_A + tid]; }
  const float Mng = Mn[g], Sng = Sn[g];
  const int wave = tid >> 5, lane = tid & 31;
  const int m0 = (wave >> 2) * 16, a0 = (wave & 3) * 16;
  const int nch = (e - s + 31) >> 5;
  for (int ch = 0; ch < nch; ++ch) {
    const int base = s + ch * 32;
    __syncthreads();
    for (int i = 0; i < 32; ++i) {
      int n = base + i;
      float x = (n < e) ? values[(size_t)n * N_D + tid] : 0.f;
      unsigned short h = f2bf(x);
      vhi[i * N_D + tid] = h;
      vlo[i * N_D + tid] = f2bf(x - bf2f(h));
    }
    __syncthreads();
    v8f c = wmma_tile_k256(vhi, vlo, whi, wlo, m0, a0, lane);
    {
      const int col = a0 + (lane & 15);
      const float bias = b_agn[col];
      const int rbase = m0 + ((lane & 16) ? 8 : 0);
#pragma unroll
      for (int v = 0; v < 8; ++v) abuf[(rbase + v) * N_A + col] = c[v] + bias;
    }
    __syncthreads();
    if (tid < 32) {                        // masked row softmax * p_n, in place
      const int n = base + tid;
      if (n < e) {
        float mx = -INFINITY;
        for (int aa = 0; aa < N_A; ++aa) {
          float x = smask[aa] ? abuf[tid * N_A + aa] : NEGV;
          mx = fmaxf(mx, x);
        }
        float sum = 0.f;
        for (int aa = 0; aa < N_A; ++aa) {
          float x = smask[aa] ? abuf[tid * N_A + aa] : NEGV;
          float p = expf(x - mx);
          abuf[tid * N_A + aa] = p;
          sum += p;
        }
        const float scale = (expf(nl[n] - Mng) / Sng) / sum;
        for (int aa = 0; aa < N_A; ++aa) abuf[tid * N_A + aa] *= scale;
      } else {
        for (int aa = 0; aa < N_A; ++aa) abuf[tid * N_A + aa] = 0.f;
      }
    }
    __syncthreads();
    if (tid < N_A) {                       // fixed-order column sum
      float acc = 0.f;
      for (int i = 0; i < 32; ++i) acc += abuf[i * N_A + tid];
      pacc[tid] += acc;
    }
  }
  __syncthreads();
  if (tid < N_A) out_pa[g * N_A + tid] = pacc[tid];
}

// K3b: per-graph: recompute QNA/QAN via WMMA, finalize p_n__a, entropy, value
__global__ __launch_bounds__(256) void k3_nq(
    const float* __restrict__ values,
    const unsigned short* __restrict__ qna_hi, const unsigned short* __restrict__ qna_lo,
    const float* __restrict__ b_qna,
    const unsigned short* __restrict__ qan_hi, const unsigned short* __restrict__ qan_lo,
    const float* __restrict__ b_qan,
    const unsigned char* __restrict__ mask, const int* __restrict__ seg_start,
    const float* __restrict__ Mm, const float* __restrict__ Sm,
    const float* __restrict__ Mu, const float* __restrict__ Su,
    const float* __restrict__ out_pa,
    float* __restrict__ pnna,              // in: NGA logits, out: p_n__a
    float* __restrict__ out_value, float* __restrict__ out_entropy)
{
  __shared__ __align__(16) unsigned short vhi[32 * N_D];
  __shared__ __align__(16) unsigned short vlo[32 * N_D];
  __shared__ float bq1[32 * N_A];
  __shared__ float bq2[32 * N_A];
  __shared__ float p0[4][64], p1[4][64], p2[4][64];
  __shared__ float tA[64], qA[64], hA[64];
  const int g = blockIdx.x;
  const int s = seg_start[g], e = seg_start[g + 1];
  const int tid = threadIdx.x;
  const int a = tid & 63, r = tid >> 6;
  const bool mk = mask[g * N_A + a] != 0;
  const float Mma = Mm[g * N_A + a], Sma = Sm[g * N_A + a];
  const float Mua = Mu[g * N_A + a], Sua = Su[g * N_A + a];
  float acc_t = 0.f, acc_q = 0.f, acc_h = 0.f;
  const int wave = tid >> 5, lane = tid & 31;
  const int m0 = (wave >> 2) * 16, a0 = (wave & 3) * 16;
  const int nch = (e - s + 31) >> 5;
  for (int ch = 0; ch < nch; ++ch) {
    const int base = s + ch * 32;
    __syncthreads();
    for (int i = 0; i < 32; ++i) {
      int n = base + i;
      float x = (n < e) ? values[(size_t)n * N_D + tid] : 0.f;
      unsigned short h = f2bf(x);
      vhi[i * N_D + tid] = h;
      vlo[i * N_D + tid] = f2bf(x - bf2f(h));
    }
    __syncthreads();
    v8f c1 = wmma_tile_k256(vhi, vlo, qna_hi, qna_lo, m0, a0, lane);
    v8f c2 = wmma_tile_k256(vhi, vlo, qan_hi, qan_lo, m0, a0, lane);
    {
      const int col = a0 + (lane & 15);
      const float b1 = b_qna[col], b2 = b_qan[col];
      const int rbase = m0 + ((lane & 16) ? 8 : 0);
#pragma unroll
      for (int v = 0; v < 8; ++v) {
        bq1[(rbase + v) * N_A + col] = c1[v] + b1;
        bq2[(rbase + v) * N_A + col] = c2[v] + b2;
      }
    }
    __syncthreads();
    for (int i = r; i < 32; i += 4) {      // thread owns column a, rows r,r+4,...
      const int n = base + i;
      if (n < e) {
        float x  = pnna[(size_t)n * N_A + a];
        float pu = expf(x - Mua) / Sua;
        pnna[(size_t)n * N_A + a] = pu;    // finalize p_n__a in place
        float pmv = expf((mk ? x : NEGV) - Mma) / Sma;
        acc_h -= pmv * logf(pmv + EPSV);
        acc_t += pu * bq1[i * N_A + a];
        acc_q += bq2[i * N_A + a];
      }
    }
  }
  __syncthreads();
  p0[r][a] = acc_t; p1[r][a] = acc_q; p2[r][a] = acc_h;
  __syncthreads();
  if (tid < 64) {
    float T = 0.f, Q = 0.f, H = 0.f;
    for (int rr = 0; rr < 4; ++rr) { T += p0[rr][tid]; Q += p1[rr][tid]; H += p2[rr][tid]; }
    tA[tid] = T; qA[tid] = Q; hA[tid] = H;
  }
  __syncthreads();
  if (tid == 0) {
    float val = 0.f, Ha = 0.f, e2 = 0.f;
    for (int aa = 0; aa < N_A; ++aa) {
      float pa = out_pa[g * N_A + aa];
      val += pa * (qA[aa] + tA[aa]);
      Ha  -= pa * logf(pa + EPSV);
      e2  += pa * hA[aa];
    }
    out_value[g]   = val;
    out_entropy[g] = Ha + e2;
  }
}

// K4: logprob[g] = log(p_a[g,a0]+EPS) + log(p_masked[n0,a0]+EPS), exact fp32 dot
__global__ __launch_bounds__(64) void k4_logprob(
    const float* __restrict__ values, const float* __restrict__ W_nga,
    const float* __restrict__ b_nga, const int* __restrict__ indices,
    const int* __restrict__ a_action, const int* __restrict__ a_node,
    const unsigned char* __restrict__ mask,
    const float* __restrict__ Mm, const float* __restrict__ Sm,
    const float* __restrict__ out_pa, float* __restrict__ out_logprob)
{
  const int g = blockIdx.x, tid = threadIdx.x;
  const int a0 = a_action[g], n0 = a_node[g];
  const int g0 = indices[n0];
  __shared__ float red[64];
  float s = 0.f;
  for (int d = tid; d < N_D; d += 64)
    s += values[(size_t)n0 * N_D + d] * W_nga[a0 * N_D + d];
  red[tid] = s;
  __syncthreads();
  for (int off = 32; off > 0; off >>= 1) {
    if (tid < off) red[tid] += red[tid + off];
    __syncthreads();
  }
  if (tid == 0) {
    float x  = red[0] + b_nga[a0];
    float xm = mask[g0 * N_A + a0] ? x : NEGV;
    float p2 = expf(xm - Mm[g0 * N_A + a0]) / Sm[g0 * N_A + a0];
    float p1 = out_pa[g * N_A + a0];
    out_logprob[g] = logf(p1 + EPSV) + logf(p2 + EPSV);
  }
}

extern "C" void kernel_launch(void* const* d_in, const int* in_sizes, int n_in,
                              void* d_out, int out_size, void* d_ws, size_t ws_size,
                              hipStream_t stream)
{
  (void)in_sizes; (void)n_in; (void)out_size; (void)ws_size;
  const float* values        = (const float*)d_in[0];
  const int*   indices       = (const int*)d_in[1];
  const int*   a_action      = (const int*)d_in[2];
  const int*   a_node        = (const int*)d_in[3];
  const unsigned char* amask = (const unsigned char*)d_in[4];  // jnp bool = 1 byte
  const int*   n_nodes       = (const int*)d_in[5];
  const float* w_node        = (const float*)d_in[6];
  const float* W_agn         = (const float*)d_in[7];
  const float* b_agn         = (const float*)d_in[8];
  const float* W_nga         = (const float*)d_in[9];
  const float* b_nga         = (const float*)d_in[10];
  const float* W_qna         = (const float*)d_in[11];
  const float* b_qna         = (const float*)d_in[12];
  const float* W_qan         = (const float*)d_in[13];
  const float* b_qan         = (const float*)d_in[14];

  float* out       = (float*)d_out;
  float* o_logprob = out;
  float* o_entropy = out + N_G;
  float* o_value   = out + 2 * N_G;
  float* o_pa      = out + 3 * N_G;
  float* o_pnna    = out + 3 * N_G + N_G * N_A;   // [N, A]

  // workspace layout (~1 MB)
  char* w = (char*)d_ws;
  size_t off = 0;
  auto take = [&](size_t bytes) { void* p = w + off; off += (bytes + 255) & ~(size_t)255; return p; };
  int*   seg    = (int*)  take((N_G + 1) * sizeof(int));
  float* nl     = (float*)take((size_t)N_NODES * sizeof(float));
  float* Mn     = (float*)take(N_G * sizeof(float));
  float* Sn     = (float*)take(N_G * sizeof(float));
  float* Mm     = (float*)take((size_t)N_G * N_A * sizeof(float));
  float* Sm     = (float*)take((size_t)N_G * N_A * sizeof(float));
  float* Mu     = (float*)take((size_t)N_G * N_A * sizeof(float));
  float* Su     = (float*)take((size_t)N_G * N_A * sizeof(float));
  unsigned short* agn_hi = (unsigned short*)take((size_t)N_A * N_D * 2);
  unsigned short* agn_lo = (unsigned short*)take((size_t)N_A * N_D * 2);
  unsigned short* nga_hi = (unsigned short*)take((size_t)N_A * N_D * 2);
  unsigned short* nga_lo = (unsigned short*)take((size_t)N_A * N_D * 2);
  unsigned short* qna_hi = (unsigned short*)take((size_t)N_A * N_D * 2);
  unsigned short* qna_lo = (unsigned short*)take((size_t)N_A * N_D * 2);
  unsigned short* qan_hi = (unsigned short*)take((size_t)N_A * N_D * 2);
  unsigned short* qan_lo = (unsigned short*)take((size_t)N_A * N_D * 2);

  k0_scan<<<1, 64, 0, stream>>>(n_nodes, seg);
  k0b_conv<<<64, 256, 0, stream>>>(W_agn, agn_hi, agn_lo);
  k0b_conv<<<64, 256, 0, stream>>>(W_nga, nga_hi, nga_lo);
  k0b_conv<<<64, 256, 0, stream>>>(W_qna, qna_hi, qna_lo);
  k0b_conv<<<64, 256, 0, stream>>>(W_qan, qan_hi, qan_lo);

  k1_nga_nl<<<N_NODES / 32, 256, 0, stream>>>(values, nga_hi, nga_lo, b_nga, w_node,
                                              o_pnna, nl);
  k2_seg<<<N_G, 256, 0, stream>>>(o_pnna, amask, seg, nl, Mm, Sm, Mu, Su, Mn, Sn);
  k3_agn<<<N_G, 256, 0, stream>>>(values, agn_hi, agn_lo, b_agn, amask, seg, nl,
                                  Mn, Sn, o_pa);
  k3_nq<<<N_G, 256, 0, stream>>>(values, qna_hi, qna_lo, b_qna, qan_hi, qan_lo, b_qan,
                                 amask, seg, Mm, Sm, Mu, Su, o_pa, o_pnna,
                                 o_value, o_entropy);
  k4_logprob<<<N_G, 64, 0, stream>>>(values, W_nga, b_nga, indices, a_action, a_node,
                                     amask, Mm, Sm, o_pa, o_logprob);
}
